// CTRNN_41154376630459
// MI455X (gfx1250) — compile-verified, matrix operand order checked by reading, not run on hardware
//
#include <hip/hip_runtime.h>
#include <stdint.h>

#define T_ 512
#define B_ 64
#define I_ 256
#define H_ 1024
#define ALPHA_ 0.1f
#define OMA_   0.9f

typedef __attribute__((ext_vector_type(16))) __bf16 v16bf;
typedef __attribute__((ext_vector_type(8)))  float  v8f;
typedef __attribute__((ext_vector_type(4)))  unsigned int v4u;
typedef __attribute__((ext_vector_type(8)))  int v8i;
typedef __attribute__((ext_vector_type(4)))  int v4i;

extern __shared__ char smem[];

__device__ __forceinline__ unsigned short f2b(float f) {
  unsigned u = __builtin_bit_cast(unsigned, f);
  u += 0x7fffu + ((u >> 16) & 1u);          // round-to-nearest-even to bf16
  return (unsigned short)(u >> 16);
}

__device__ __forceinline__ v8f wmma_bf16(v16bf a, v16bf b, v8f c) {
  return __builtin_amdgcn_wmma_f32_16x16x32_bf16(false, a, false, b, (short)0, c,
                                                 false, false);
}

__device__ __forceinline__ unsigned lds_lo(const void* p) {
  // flat LDS pointer: low 32 bits == LDS byte address
  return (unsigned)(uintptr_t)p;
}

// Build a minimal 2D Tensor-DMA descriptor and issue TENSOR_LOAD_TO_LDS.
// dims/tiles in elements (data_size = 2 bytes / bf16), stride0 in elements.
// 6-arg builtin form: (g0 v4u, g1 v8i, g2 v4i, g3 v4i, v8i, i32 cpol).
__device__ __forceinline__ void tdm_load_2d(unsigned lds_addr, const void* gptr,
                                            unsigned dim0, unsigned dim1,
                                            unsigned tile0, unsigned tile1,
                                            unsigned stride0) {
  unsigned long long ga = (unsigned long long)(uintptr_t)gptr;
  v4u g0 = { 1u,                                    // count=1, user descriptor
             lds_addr,                              // lds_addr [63:32]
             (unsigned)ga,                          // global_addr [95:64]
             (unsigned)((ga >> 32) & 0x1FFFFFFull)  // global_addr [120:96]
             | (2u << 30) };                        // type = 2 ("image")
  v8i g1 = { (int)(1u << 16),                               // data_size=1 (2B)
             (int)(dim0 << 16),                             // tensor_dim0 lo16
             (int)((dim0 >> 16) | (dim1 << 16)),            // dim0 hi / dim1 lo
             (int)((dim1 >> 16) | (tile0 << 16)),           // dim1 hi / tile_dim0
             (int)(tile1 & 0xffffu),                        // tile_dim1 (tile_dim2=0)
             (int)stride0,                                  // tensor_dim0_stride lo32
             0, 0 };
  v4i z4 = {0, 0, 0, 0};
  v8i z8 = {0, 0, 0, 0, 0, 0, 0, 0};
  __builtin_amdgcn_tensor_load_to_lds(g0, g1, z4, z4, z8, 0);
}

// ---------------------------------------------------------------------------
// Kernel 0: convert W -> bf16 (Wx, Wh), h0 -> bf16 ping buffer, zero barriers
// ---------------------------------------------------------------------------
__global__ void ctrnn_convert(const float* __restrict__ W,
                              const float* __restrict__ h0,
                              unsigned short* __restrict__ Wxb,
                              unsigned short* __restrict__ Whb,
                              unsigned short* __restrict__ hping,
                              int* __restrict__ cnt) {
  const long NW = (long)H_ * (I_ + H_);   // 1,310,720
  const long NH = (long)B_ * H_;          //    65,536
  long i = (long)blockIdx.x * blockDim.x + threadIdx.x;
  if (i < NW) {
    int h = (int)(i / (I_ + H_));
    int c = (int)(i % (I_ + H_));
    unsigned short v = f2b(W[i]);
    if (c < I_) Wxb[h * I_ + c] = v;
    else        Whb[h * H_ + (c - I_)] = v;
  } else if (i < NW + NH) {
    long j = i - NW;
    hping[j] = f2b(h0[j]);
  } else if (i < NW + NH + 4) {
    cnt[i - NW - NH] = 0;
  }
}

// ---------------------------------------------------------------------------
// Kernel 1: xa = x @ Wx^T + b  (M=32768, N=1024, K=256), bf16 WMMA, f32 out.
// Writes xa into the y region of d_out (overwritten in place by the scan).
// Block = 256 threads (8 waves), tile 64x64, dynamic LDS = 64 KB.
// ---------------------------------------------------------------------------
__global__ void ctrnn_xa_gemm(const float* __restrict__ x,
                              const unsigned short* __restrict__ Wxb,
                              const float* __restrict__ bias,
                              float* __restrict__ out) {
  const int n0 = blockIdx.x * 64;
  const int m0 = blockIdx.y * 64;
  const int tid = threadIdx.x;

  unsigned short* As = (unsigned short*)smem;                 // [64][256] bf16
  unsigned short* Bs = (unsigned short*)(smem + 64 * 256 * 2);

  {
    int row = tid >> 2;
    int kc  = (tid & 3) * 64;
    const float4* src = (const float4*)(x + (size_t)(m0 + row) * I_ + kc);
    unsigned short* dst = As + row * I_ + kc;
    #pragma unroll
    for (int i = 0; i < 16; ++i) {
      float4 v = src[i];
      dst[i * 4 + 0] = f2b(v.x); dst[i * 4 + 1] = f2b(v.y);
      dst[i * 4 + 2] = f2b(v.z); dst[i * 4 + 3] = f2b(v.w);
    }
    const uint4* ws = (const uint4*)(Wxb + (size_t)(n0 + row) * I_ + kc);
    uint4* wd = (uint4*)(Bs + row * I_ + kc);
    #pragma unroll
    for (int i = 0; i < 8; ++i) wd[i] = ws[i];
  }
  __syncthreads();

  const int w = tid >> 5, lane = tid & 31;
  const int nl = lane & 15, hi = lane >> 4;
  const int mt0 = w >> 2, nt = w & 3;   // wave handles C tiles (mt0,nt) & (mt0+2,nt)

  v8f c0 = {}, c1 = {};
  #pragma unroll
  for (int kk = 0; kk < 8; ++kk) {
    const v16bf b  = *(const v16bf*)(Bs + (nt * 16 + nl) * I_ + kk * 32 + hi * 16);
    const v16bf a0 = *(const v16bf*)(As + (mt0 * 16 + nl) * I_ + kk * 32 + hi * 16);
    const v16bf a1 = *(const v16bf*)(As + ((mt0 + 2) * 16 + nl) * I_ + kk * 32 + hi * 16);
    c0 = wmma_bf16(a0, b, c0);
    c1 = wmma_bf16(a1, b, c1);
  }

  const int col = n0 + nt * 16 + nl;
  const float bv = bias[col];
  #pragma unroll
  for (int r = 0; r < 8; ++r) {
    int m = r + 8 * hi;
    out[(size_t)(m0 + mt0 * 16 + m) * H_ + col]       = c0[r] + bv;
    out[(size_t)(m0 + (mt0 + 2) * 16 + m) * H_ + col] = c1[r] + bv;
  }
}

// ---------------------------------------------------------------------------
// Kernel 2: persistent recurrence scan.
// 64 WGs = 4 batch-groups (16 rows each) x 16 H-slices (64 cols each).
// Wh slice (64x1024 bf16, 128 KB) resident in LDS via TDM; per step the
// group's h tile (16x1024 bf16, 32 KB) is TDM-loaded from an L2 ping-pong
// buffer. f32 state lives in the WMMA accumulator layout across all steps.
// Block = 128 threads (4 waves), dynamic LDS = 160 KB.
// ---------------------------------------------------------------------------
__global__ void ctrnn_scan(const float* __restrict__ h0,
                           float* __restrict__ out,          // xa in, y out
                           const unsigned short* __restrict__ Whb,
                           unsigned short* __restrict__ hb0,
                           unsigned short* __restrict__ hb1,
                           int* __restrict__ cnt) {
  const int wg = blockIdx.x;
  const int g  = wg >> 4;     // batch group 0..3
  const int s  = wg & 15;     // H slice 0..15
  const int tid = threadIdx.x;
  const int w = tid >> 5, lane = tid & 31;
  const int nl = lane & 15, hi = lane >> 4;

  unsigned short* Wh_lds = (unsigned short*)smem;                  // 64*1024 bf16
  unsigned short* hA     = (unsigned short*)(smem + 64 * H_ * 2);  // 16*1024 bf16

  // Preload this slice's Wh rows [s*64, s*64+64) into LDS (once).
  if (tid < 32) {
    tdm_load_2d(lds_lo(Wh_lds), Whb + (size_t)s * 64 * H_,
                H_, 64, H_, 64, H_);
    __builtin_amdgcn_s_wait_tensorcnt(0);
  }
  __syncthreads();

  const int colbase = s * 64 + w * 16;
  const int col = colbase + nl;

  // f32 hidden state in C/D accumulator layout: vgpr r, lane -> (m=r+8*hi, n=nl)
  float hold[8];
  #pragma unroll
  for (int r = 0; r < 8; ++r) {
    int m = r + 8 * hi;
    hold[r] = h0[(size_t)(g * 16 + m) * H_ + col];
  }

  int* mycnt = cnt + g;

  for (int t = 0; t < T_; ++t) {
    // ---- group barrier: all 16 WGs of this batch group ----
    __threadfence();
    __syncthreads();
    if (tid == 0) {
      __hip_atomic_fetch_add(mycnt, 1, __ATOMIC_RELEASE, __HIP_MEMORY_SCOPE_AGENT);
      int target = 16 * (t + 1);
      while (__hip_atomic_load(mycnt, __ATOMIC_ACQUIRE, __HIP_MEMORY_SCOPE_AGENT)
             < target)
        __builtin_amdgcn_s_sleep(1);
    }
    __syncthreads();

    const unsigned short* rd = (t & 1) ? hb1 : hb0;
    unsigned short*       wr = (t & 1) ? hb0 : hb1;

    // ---- TDM: stage this group's 16x1024 bf16 h tile into LDS ----
    if (tid < 32) {
      tdm_load_2d(lds_lo(hA), rd + (size_t)g * 16 * H_,
                  H_, 16, H_, 16, H_);
      __builtin_amdgcn_s_wait_tensorcnt(0);
    }
    __syncthreads();

    // ---- C = h_tile (16xK) x Wh_slice^T (Kx16) over K=1024 ----
    v8f c = {};
    #pragma unroll 8
    for (int kk = 0; kk < 32; ++kk) {
      const v16bf a = *(const v16bf*)(hA + nl * H_ + kk * 32 + hi * 16);
      const v16bf b = *(const v16bf*)(Wh_lds + (w * 16 + nl) * H_ + kk * 32 + hi * 16);
      c = wmma_bf16(a, b, c);
    }

    // ---- elementwise leaky-ReLU update + y write + bf16 h publish ----
    #pragma unroll
    for (int r = 0; r < 8; ++r) {
      int m = r + 8 * hi;
      int brow = g * 16 + m;
      size_t adr = ((size_t)t * B_ + brow) * H_ + col;
      float xa = out[adr];
      float hn = OMA_ * hold[r] + ALPHA_ * (xa + c[r]);
      hn = fmaxf(hn, 0.0f);
      out[adr] = hn;           // y[t]
      hold[r] = hn;
      wr[(size_t)brow * H_ + col] = f2b(hn);
    }
  }

  // h_final
  #pragma unroll
  for (int r = 0; r < 8; ++r) {
    int m = r + 8 * hi;
    out[(size_t)T_ * B_ * H_ + (size_t)(g * 16 + m) * H_ + col] = hold[r];
  }
}

// ---------------------------------------------------------------------------
extern "C" void kernel_launch(void* const* d_in, const int* in_sizes, int n_in,
                              void* d_out, int out_size, void* d_ws, size_t ws_size,
                              hipStream_t stream) {
  (void)in_sizes; (void)n_in; (void)out_size; (void)ws_size;
  const float* x    = (const float*)d_in[0];   // (T,B,I)
  const float* h0   = (const float*)d_in[1];   // (B,H)
  const float* W    = (const float*)d_in[2];   // (H, I+H)
  const float* bias = (const float*)d_in[3];   // (H,)
  float* out = (float*)d_out;                  // y (T,B,H) ++ h_final (B,H)

  char* ws = (char*)d_ws;
  unsigned short* Wxb = (unsigned short*)(ws);                 // 1024*256 bf16
  unsigned short* Whb = (unsigned short*)(ws + 524288);        // 1024*1024 bf16
  unsigned short* hb0 = (unsigned short*)(ws + 2621440);       // 64*1024 bf16
  unsigned short* hb1 = (unsigned short*)(ws + 2752512);       // 64*1024 bf16
  int*            cnt = (int*)(ws + 2883584);                  // 4 counters

  // 0) convert weights/h0 to bf16, reset barrier counters
  {
    long total = (long)H_ * (I_ + H_) + (long)B_ * H_ + 4;
    int grid = (int)((total + 255) / 256);
    ctrnn_convert<<<grid, 256, 0, stream>>>(W, h0, Wxb, Whb, hb0, cnt);
  }
  // 1) xa = x @ Wx^T + b  -> y region of d_out
  {
    dim3 grid(H_ / 64, (T_ * B_) / 64);      // (16, 512)
    ctrnn_xa_gemm<<<grid, 256, 64 * 1024, stream>>>(x, Wxb, bias, out);
  }
  // 2) persistent recurrent scan
  {
    ctrnn_scan<<<64, 128, 160 * 1024, stream>>>(h0, out, Whb, hb0, hb1, cnt);
  }
}
